// Attention_62294205661445
// MI455X (gfx1250) — compile-verified, matrix operand order checked by reading, not run on hardware
//
#include <hip/hip_runtime.h>

typedef __attribute__((ext_vector_type(16))) __bf16 v16bf;
typedef __attribute__((ext_vector_type(8)))  float  v8f;

constexpr int B_ = 4, S_ = 2048, D_ = 1152, H_ = 16, KVH_ = 4, HD_ = 64, SW_ = 1792;
constexpr int E_ = H_ * HD_;      // 1024
constexpr int KVE_ = KVH_ * HD_;  // 256
constexpr int M_ = B_ * S_;       // 8192
// 1/sqrt(HD) * log2(e): softmax runs in the exp2 domain (v_exp_f32 with no extra mul)
constexpr float QSCALE = 0.125f * 1.4426950408889634f;

// ---------------- helpers ----------------

__device__ __forceinline__ __bf16 to_bf16(float f) { return (__bf16)f; }

union FragU { unsigned u[8]; v16bf v; };

// Load a 16x32 bf16 WMMA fragment from a row-major matrix.
// Works for both A (lane%16 = M row) and B (lane%16 = N column reading rows of W / Vt),
// per the CDNA5 ISA 16-bit layout: VGPR j<4 -> K = g*8 + 2j(+1), j>=4 -> K = 16 + g*8 + 2(j-4)(+1),
// g = lane/16.  Requires ld and kbase even (4-byte alignment).
__device__ __forceinline__ v16bf load_frag(const __bf16* base, int ld, int row, int kbase) {
  int lane = threadIdx.x & 31;
  int g = lane >> 4;
  const unsigned* p =
      (const unsigned*)(base + (size_t)(row + (lane & 15)) * ld + kbase + g * 8);
  FragU f;
#pragma unroll
  for (int j = 0; j < 4; ++j) {
    f.u[j]     = p[j];      // K = g*8 + 2j, 2j+1
    f.u[4 + j] = p[8 + j];  // K = 16 + g*8 + 2j, 2j+1
  }
  return f.v;
}

__device__ __forceinline__ v8f wmma_bf16(v16bf a, v16bf b, v8f c) {
  return __builtin_amdgcn_wmma_f32_16x16x32_bf16(false, a, false, b, (short)0, c,
                                                 false, false);
}

// C/D layout: lane holds (M = r + 8*(lane/16), N = lane%16) in VGPR r.
__device__ __forceinline__ void store_tile(float* C, int ldc, int row, int col, v8f acc) {
  int lane = threadIdx.x & 31;
  int n = lane & 15, g = lane >> 4;
#pragma unroll
  for (int r = 0; r < 8; ++r)
    C[(size_t)(row + r + 8 * g) * ldc + col + n] = acc[r];
}

// ---------------- fp32 -> bf16 convert ----------------

__global__ void cvt_kernel(const float* __restrict__ in, __bf16* __restrict__ out, int n) {
  int i = blockIdx.x * blockDim.x + threadIdx.x;
  if (i < n) out[i] = to_bf16(in[i]);
}

// ---------------- generic GEMM: C[M,N] = A[M,K] * W[N,K]^T ----------------
// block = 128 threads (4 waves), each wave computes a 32x64 tile
// (2 A-fragments reuse every B-fragment -> ~48 B loaded per WMMA).

__global__ void gemm_bf16(const __bf16* __restrict__ A, const __bf16* __restrict__ W,
                          float* __restrict__ C, int M, int N, int K) {
  int wave = threadIdx.x >> 5;
  int row0 = blockIdx.y * 128 + wave * 32;
  int col0 = blockIdx.x * 64;
  const v8f z = {0, 0, 0, 0, 0, 0, 0, 0};
  v8f acc[2][4];
#pragma unroll
  for (int mb = 0; mb < 2; ++mb)
#pragma unroll
    for (int c = 0; c < 4; ++c) acc[mb][c] = z;

  for (int k = 0; k < K; k += 32) {
    v16bf a0 = load_frag(A, K, row0, k);
    v16bf a1 = load_frag(A, K, row0 + 16, k);
#pragma unroll
    for (int c = 0; c < 4; ++c) {
      v16bf w = load_frag(W, K, col0 + 16 * c, k);
      acc[0][c] = wmma_bf16(a0, w, acc[0][c]);
      acc[1][c] = wmma_bf16(a1, w, acc[1][c]);
    }
  }
#pragma unroll
  for (int mb = 0; mb < 2; ++mb)
#pragma unroll
    for (int c = 0; c < 4; ++c)
      store_tile(C, N, row0 + 16 * mb, col0 + 16 * c, acc[mb][c]);
}

// ---------------- RoPE Q: rotate, pre-scale by QSCALE, emit bf16 [B][H][S][64] ----------

__global__ void rope_q_kernel(const float* __restrict__ qf, const float* __restrict__ fc,
                              const float* __restrict__ fs, __bf16* __restrict__ qb) {
  int idx = blockIdx.x * blockDim.x + threadIdx.x;  // B*S*H*32 threads
  int i = idx & 31;
  int h = (idx >> 5) & 15;
  int s = (idx >> 9) & 2047;
  int b = idx >> 20;
  size_t src = ((size_t)(b * S_ + s)) * E_ + h * HD_ + 2 * i;
  float a = qf[src], bb = qf[src + 1];
  float c = fc[s * 32 + i], sn = fs[s * 32 + i];
  float ra = (a * c - bb * sn) * QSCALE;
  float rb = (a * sn + bb * c) * QSCALE;
  size_t o = ((size_t)((b * H_ + h) * S_ + s)) * HD_ + 2 * i;
  qb[o] = to_bf16(ra);
  qb[o + 1] = to_bf16(rb);
}

// ---------------- RoPE K + V reorg + KV cache scatter ----------------
// kbo: bf16 [B][KVH][S][64] (post-rope)
// vto: bf16 [B][KVH][64][S] (transposed V for contiguous PV B-fragments)
// new_ck/new_cv: fp32 (B, SW, KVH, HD); slot = s % SW for s >= S-SW (=256)

__global__ void rope_kv_kernel(const float* __restrict__ kf, const float* __restrict__ vf,
                               const float* __restrict__ fc, const float* __restrict__ fs,
                               __bf16* __restrict__ kbo, __bf16* __restrict__ vto,
                               float* __restrict__ new_ck, float* __restrict__ new_cv) {
  int idx = blockIdx.x * blockDim.x + threadIdx.x;  // B*S*KVH*32 threads
  int i = idx & 31;
  int kvh = (idx >> 5) & 3;
  int s = (idx >> 7) & 2047;
  int b = idx >> 18;
  size_t src = ((size_t)(b * S_ + s)) * KVE_ + kvh * HD_ + 2 * i;
  float ka = kf[src], kb = kf[src + 1];
  float va = vf[src], vb = vf[src + 1];
  float c = fc[s * 32 + i], sn = fs[s * 32 + i];
  float ra = ka * c - kb * sn;
  float rb = ka * sn + kb * c;
  size_t ko = ((size_t)((b * KVH_ + kvh) * S_ + s)) * HD_ + 2 * i;
  kbo[ko] = to_bf16(ra);
  kbo[ko + 1] = to_bf16(rb);
  size_t vbase = ((size_t)(b * KVH_ + kvh)) * HD_ * S_;
  vto[vbase + (size_t)(2 * i) * S_ + s] = to_bf16(va);
  vto[vbase + (size_t)(2 * i + 1) * S_ + s] = to_bf16(vb);
  if (s >= S_ - SW_) {
    int slot = s >= SW_ ? s - SW_ : s;  // s % SW
    size_t co = ((size_t)((b * SW_ + slot) * KVH_ + kvh)) * HD_ + 2 * i;
    new_ck[co] = ra;
    new_ck[co + 1] = rb;
    new_cv[co] = va;
    new_cv[co + 1] = vb;
  }
}

// ---------------- flash attention with sliding window ----------------
// grid = (S/128, H, B), block = 128 (4 waves); each wave owns 32 query rows
// (two 16-row M-blocks), so every K/V fragment feeds 2 score / 2 PV WMMAs.

__global__ void attn_kernel(const __bf16* __restrict__ qb, const __bf16* __restrict__ kb,
                            const __bf16* __restrict__ vt, __bf16* __restrict__ attn) {
  int wave = threadIdx.x >> 5;
  int lane = threadIdx.x & 31;
  int h = blockIdx.y, b = blockIdx.z;
  int i0 = blockIdx.x * 128 + wave * 32;
  int kvh = h >> 2;  // GQA: 4 query heads per kv head

  const __bf16* qbh = qb + ((size_t)(b * H_ + h) * S_) * HD_;
  const __bf16* kbh = kb + ((size_t)(b * KVH_ + kvh) * S_) * HD_;
  const __bf16* vth = vt + ((size_t)(b * KVH_ + kvh)) * HD_ * S_;

  v16bf qa[2][2];
#pragma unroll
  for (int mb = 0; mb < 2; ++mb) {
    qa[mb][0] = load_frag(qbh, HD_, i0 + 16 * mb, 0);
    qa[mb][1] = load_frag(qbh, HD_, i0 + 16 * mb, 32);
  }

  const v8f z = {0, 0, 0, 0, 0, 0, 0, 0};
  v8f o[2][4];
  float mst[2][8], lst[2][8];
#pragma unroll
  for (int mb = 0; mb < 2; ++mb) {
#pragma unroll
    for (int c = 0; c < 4; ++c) o[mb][c] = z;
#pragma unroll
    for (int r = 0; r < 8; ++r) { mst[mb][r] = -1e30f; lst[mb][r] = 0.f; }
  }

  __shared__ __bf16 psh[4][2][16 * 32];
  int g = lane >> 4, n = lane & 15;

  int jstart = i0 - (SW_ - 1);
  if (jstart < 0) jstart = 0;
  jstart &= ~31;
  int jend = i0 + 31;

  for (int j0 = jstart; j0 <= jend; j0 += 32) {
    v16bf k0l = load_frag(kbh, HD_, j0, 0);
    v16bf k0h = load_frag(kbh, HD_, j0, 32);
    v16bf k1l = load_frag(kbh, HD_, j0 + 16, 0);
    v16bf k1h = load_frag(kbh, HD_, j0 + 16, 32);

#pragma unroll
    for (int mb = 0; mb < 2; ++mb) {
      // scores (exp2 domain): 16 queries x 32 keys, f32 accumulate
      v8f s0 = wmma_bf16(qa[mb][0], k0l, z);
      s0 = wmma_bf16(qa[mb][1], k0h, s0);
      v8f s1 = wmma_bf16(qa[mb][0], k1l, z);
      s1 = wmma_bf16(qa[mb][1], k1h, s1);
      __bf16* pw = psh[wave][mb];
#pragma unroll
      for (int r = 0; r < 8; ++r) {
        int i = i0 + 16 * mb + r + 8 * g;
        int ja = j0 + n, jb = j0 + 16 + n;
        bool a0 = (ja <= i) && (i - ja < SW_);
        bool a1 = (jb <= i) && (i - jb < SW_);
        float p0 = a0 ? s0[r] : -1e30f;
        float p1 = a1 ? s1[r] : -1e30f;
        float mx = fmaxf(p0, p1);
        mx = fmaxf(mx, __shfl_xor(mx, 1, 16));
        mx = fmaxf(mx, __shfl_xor(mx, 2, 16));
        mx = fmaxf(mx, __shfl_xor(mx, 4, 16));
        mx = fmaxf(mx, __shfl_xor(mx, 8, 16));
        float mnew = fmaxf(mst[mb][r], mx);
        float e0 = a0 ? exp2f(p0 - mnew) : 0.f;  // explicit 0 keeps fully-masked blocks safe
        float e1 = a1 ? exp2f(p1 - mnew) : 0.f;
        float corr = exp2f(mst[mb][r] - mnew);
        mst[mb][r] = mnew;
        float rs = e0 + e1;
        rs += __shfl_xor(rs, 1, 16);
        rs += __shfl_xor(rs, 2, 16);
        rs += __shfl_xor(rs, 4, 16);
        rs += __shfl_xor(rs, 8, 16);
        lst[mb][r] = lst[mb][r] * corr + rs;
        o[mb][0][r] *= corr; o[mb][1][r] *= corr;
        o[mb][2][r] *= corr; o[mb][3][r] *= corr;
        // bounce P tile (16x32) through per-wave LDS to reshape C-layout -> A-fragment
        pw[(r + 8 * g) * 32 + n] = to_bf16(e0);
        pw[(r + 8 * g) * 32 + 16 + n] = to_bf16(e1);
      }
    }
    asm volatile("s_wait_dscnt 0" ::: "memory");  // wave-internal LDS RAW fence
    // PV: B fragments are contiguous rows of Vt (column d of V), ld = S; reused by both M-blocks
    v16bf v0 = load_frag(vth, S_, 0, j0);
    v16bf v1 = load_frag(vth, S_, 16, j0);
    v16bf v2 = load_frag(vth, S_, 32, j0);
    v16bf v3 = load_frag(vth, S_, 48, j0);
#pragma unroll
    for (int mb = 0; mb < 2; ++mb) {
      v16bf pa = load_frag(psh[wave][mb], 32, 0, 0);
      o[mb][0] = wmma_bf16(pa, v0, o[mb][0]);
      o[mb][1] = wmma_bf16(pa, v1, o[mb][1]);
      o[mb][2] = wmma_bf16(pa, v2, o[mb][2]);
      o[mb][3] = wmma_bf16(pa, v3, o[mb][3]);
    }
  }

#pragma unroll
  for (int mb = 0; mb < 2; ++mb)
#pragma unroll
    for (int r = 0; r < 8; ++r) {
      float inv = 1.0f / lst[mb][r];
      int i = i0 + 16 * mb + r + 8 * g;
      size_t rowbase = ((size_t)(b * S_ + i)) * E_ + h * HD_;
      attn[rowbase + n]      = to_bf16(o[mb][0][r] * inv);
      attn[rowbase + 16 + n] = to_bf16(o[mb][1][r] * inv);
      attn[rowbase + 32 + n] = to_bf16(o[mb][2][r] * inv);
      attn[rowbase + 48 + n] = to_bf16(o[mb][3][r] * inv);
    }
}

// ---------------- host launcher ----------------

extern "C" void kernel_launch(void* const* d_in, const int* in_sizes, int n_in,
                              void* d_out, int out_size, void* d_ws, size_t ws_size,
                              hipStream_t stream) {
  const float* x  = (const float*)d_in[0];
  const float* wq = (const float*)d_in[1];
  const float* wk = (const float*)d_in[2];
  const float* wv = (const float*)d_in[3];
  const float* wo = (const float*)d_in[4];
  const float* fc = (const float*)d_in[5];
  const float* fs = (const float*)d_in[6];
  // d_in[7..10] (mask, cache_k, cache_v, positions) are not needed:
  // mask/window is computed analytically and the cache is fully overwritten.

  float* out    = (float*)d_out;
  float* new_ck = out + (size_t)B_ * S_ * D_;
  float* new_cv = new_ck + (size_t)B_ * SW_ * KVH_ * HD_;

  char* ws = (char*)d_ws;
  size_t off = 0;
  auto take = [&](size_t bytes) -> char* {
    char* p = ws + off;
    off += (bytes + 255) & ~(size_t)255;
    return p;
  };
  __bf16* xb   = (__bf16*)take((size_t)M_ * D_ * 2);
  __bf16* wqb  = (__bf16*)take((size_t)E_ * D_ * 2);
  __bf16* wkb  = (__bf16*)take((size_t)KVE_ * D_ * 2);
  __bf16* wvb  = (__bf16*)take((size_t)KVE_ * D_ * 2);
  __bf16* wob  = (__bf16*)take((size_t)D_ * E_ * 2);
  float*  qf   = (float*)take((size_t)M_ * E_ * 4);
  float*  kf   = (float*)take((size_t)M_ * KVE_ * 4);
  float*  vf   = (float*)take((size_t)M_ * KVE_ * 4);
  __bf16* qbf  = (__bf16*)take((size_t)M_ * E_ * 2);
  __bf16* kbf  = (__bf16*)take((size_t)M_ * KVE_ * 2);
  __bf16* vtb  = (__bf16*)take((size_t)M_ * KVE_ * 2);
  __bf16* attb = (__bf16*)take((size_t)M_ * E_ * 2);

  const int T = 256;
  cvt_kernel<<<(M_ * D_ + T - 1) / T, T, 0, stream>>>(x, xb, M_ * D_);
  cvt_kernel<<<(E_ * D_ + T - 1) / T, T, 0, stream>>>(wq, wqb, E_ * D_);
  cvt_kernel<<<(KVE_ * D_ + T - 1) / T, T, 0, stream>>>(wk, wkb, KVE_ * D_);
  cvt_kernel<<<(KVE_ * D_ + T - 1) / T, T, 0, stream>>>(wv, wvb, KVE_ * D_);
  cvt_kernel<<<(D_ * E_ + T - 1) / T, T, 0, stream>>>(wo, wob, D_ * E_);

  gemm_bf16<<<dim3(E_ / 64, M_ / 128), 128, 0, stream>>>(xb, wqb, qf, M_, E_, D_);
  gemm_bf16<<<dim3(KVE_ / 64, M_ / 128), 128, 0, stream>>>(xb, wkb, kf, M_, KVE_, D_);
  gemm_bf16<<<dim3(KVE_ / 64, M_ / 128), 128, 0, stream>>>(xb, wvb, vf, M_, KVE_, D_);

  rope_q_kernel<<<(B_ * S_ * H_ * 32) / T, T, 0, stream>>>(qf, fc, fs, qbf);
  rope_kv_kernel<<<(B_ * S_ * KVH_ * 32) / T, T, 0, stream>>>(kf, vf, fc, fs, kbf, vtb,
                                                              new_ck, new_cv);

  attn_kernel<<<dim3(S_ / 128, H_, B_), 128, 0, stream>>>(qbf, kbf, vtb, attb);

  gemm_bf16<<<dim3(D_ / 64, M_ / 128), 128, 0, stream>>>(attb, wob, out, M_, D_, E_);
}
